// cudaI_45810121179239
// MI455X (gfx1250) — compile-verified, matrix operand order checked by reading, not run on hardware
//
#include <hip/hip_runtime.h>

typedef _Float16 v16h __attribute__((ext_vector_type(16)));
typedef _Float16 v8h  __attribute__((ext_vector_type(8)));
typedef _Float16 v4h  __attribute__((ext_vector_type(4)));
typedef float    v8f  __attribute__((ext_vector_type(8)));
typedef float    v4f  __attribute__((ext_vector_type(4)));

#define DDIM 512   // feature dim
#define BT   128   // block tile edge (rows of ex and of ey)
#define KC   32    // k-chunk staged in LDS per iteration
#define NKC  (DDIM / KC)
#define LDH  40    // padded LDS row stride in halves (80B: 16B aligned, 64-bank conflict free)

// ---- monotone float <-> u32 encoding so atomicMax(u32) orders floats correctly ----
__device__ __forceinline__ unsigned encode_f32(float f) {
  unsigned u = __float_as_uint(f);
  return (u & 0x80000000u) ? ~u : (u | 0x80000000u);
}
__device__ __forceinline__ float decode_f32(unsigned e) {
  unsigned u = (e & 0x80000000u) ? (e ^ 0x80000000u) : ~e;
  return __uint_as_float(u);
}

// Load one 16x32 f16 WMMA fragment from LDS (ISA 7.12.2 16-bit A/B layout):
// lane group g=0 holds K 0..7 and 16..23, g=1 holds K 8..15 and 24..31.
__device__ __forceinline__ v16h ld_frag(const _Float16* p) {
  union { v16h v; v8h h[2]; } f;
  f.h[0] = *(const v8h*)(p);        // K base + {0..7}
  f.h[1] = *(const v8h*)(p + 16);   // K base + 16 + {0..7}
  return f.v;
}

__device__ __forceinline__ v8f wmma16(v16h a, v16h b, v8f c) {
  return __builtin_amdgcn_wmma_f32_16x16x32_f16(false, a, false, b, (short)0, c,
                                                false, false);
}

// ---- CDNA5 async global->LDS copy (VFLAT GLOBAL_LOAD_ASYNC_TO_LDS_B128, ASYNCcnt) ----
__device__ __forceinline__ unsigned to_lds_addr(const void* p) {
  // generic pointer to LDS: addr[31:0] is the LDS byte address (ISA 10.2 aperture rules)
  return (unsigned)(unsigned long long)p;
}
__device__ __forceinline__ void async_copy_b128(unsigned lds_byte_addr, const void* gaddr) {
  asm volatile("global_load_async_to_lds_b128 %0, %1, off"
               :: "v"(lds_byte_addr), "v"((unsigned long long)(size_t)gaddr)
               : "memory");
}
__device__ __forceinline__ void wait_async_le8() {
  asm volatile("s_wait_asynccnt 0x8" ::: "memory");
}
__device__ __forceinline__ void wait_async_0() {
  asm volatile("s_wait_asynccnt 0x0" ::: "memory");
}

// ---------------------------------------------------------------------------
// PATH A prep: normalize rows once, store hi/lo f16 split arrays row-major,
// and re-init rowmax every call. One block (128 threads) per row.
// ---------------------------------------------------------------------------
__global__ void prep_kernel(const float* __restrict__ ex,
                            const float* __restrict__ ey,
                            _Float16* __restrict__ Xh, _Float16* __restrict__ Xl,
                            _Float16* __restrict__ Yh, _Float16* __restrict__ Yl,
                            unsigned* __restrict__ rowmax, int Nx, int Ny) {
  int b = blockIdx.x;
  const float* src;
  _Float16 *dh, *dl;
  int row;
  if (b < Nx) { src = ex; row = b;      dh = Xh; dl = Xl; }
  else        { src = ey; row = b - Nx; dh = Yh; dl = Yl; }

  v4f v = *(const v4f*)(src + (size_t)row * DDIM + threadIdx.x * 4);
  float ss = v[0]*v[0] + v[1]*v[1] + v[2]*v[2] + v[3]*v[3];
#pragma unroll
  for (int m = 16; m; m >>= 1) ss += __shfl_xor(ss, m, 32);

  __shared__ float red[4];
  int lane = threadIdx.x & 31, wid = threadIdx.x >> 5;
  if (lane == 0) red[wid] = ss;
  __syncthreads();
  float inv = 1.0f / fmaxf(sqrtf(red[0] + red[1] + red[2] + red[3]), 1e-20f);

  v4h h, l;
#pragma unroll
  for (int t = 0; t < 4; ++t) {
    float x = v[t] * inv;
    _Float16 hi = (_Float16)x;
    h[t] = hi;
    l[t] = (_Float16)(x - (float)hi);
  }
  *(v4h*)(dh + (size_t)row * DDIM + threadIdx.x * 4) = h;
  *(v4h*)(dl + (size_t)row * DDIM + threadIdx.x * 4) = l;
  if (threadIdx.x == 0 && b < Nx) rowmax[row] = 0u;
}

// ---------------------------------------------------------------------------
// PATH A GEMM: pure async-staged WMMA loop.
//   staging: global_load_async_to_lds_b128 (8 ops/wave/chunk), double-buffered,
//   inner loop: ds_load_b128 fragments + 24 v_wmma per K=32 step.
// grid = (Ny/BT, Nx/BT), block = 256 threads = 8 wave32s
// ---------------------------------------------------------------------------
__global__ __launch_bounds__(256)
void simmax_async_kernel(const _Float16* __restrict__ Xh, const _Float16* __restrict__ Xl,
                         const _Float16* __restrict__ Yh, const _Float16* __restrict__ Yl,
                         unsigned* __restrict__ rowmax, int Nx, int Ny) {
  __shared__ _Float16 Ah[2][BT][LDH];
  __shared__ _Float16 Al[2][BT][LDH];
  __shared__ _Float16 Bh[2][BT][LDH];
  __shared__ _Float16 Bl[2][BT][LDH];

  const int tid   = threadIdx.x;
  const int rowA0 = blockIdx.y * BT;
  const int rowB0 = blockIdx.x * BT;

  // async-copy map: 4 lanes x 16B per row segment, 64 rows per pass, 2 passes
  const int rowp = tid >> 2;         // 0..63
  const int c16  = (tid & 3) << 3;   // half offset: 0,8,16,24 (16B units)

  // wave layout
  const int lane = tid & 31;
  const int g    = lane >> 4;
  const int qn   = lane & 15;
  const int wid  = tid >> 5;
  const int wm   = wid & 3;          // rows 32*wm .. +31
  const int wn   = wid >> 2;         // cols 64*wn .. +63

  auto issue_chunk = [&](int buf, int kc) {
#pragma unroll
    for (int h = 0; h < 2; ++h) {
      const int r = rowp + 64 * h;
      async_copy_b128(to_lds_addr(&Ah[buf][r][c16]), Xh + (size_t)(rowA0 + r) * DDIM + kc + c16);
      async_copy_b128(to_lds_addr(&Al[buf][r][c16]), Xl + (size_t)(rowA0 + r) * DDIM + kc + c16);
      async_copy_b128(to_lds_addr(&Bh[buf][r][c16]), Yh + (size_t)(rowB0 + r) * DDIM + kc + c16);
      async_copy_b128(to_lds_addr(&Bl[buf][r][c16]), Yl + (size_t)(rowB0 + r) * DDIM + kc + c16);
    }
  };

  v8f acc[2][4] = {};

  issue_chunk(0, 0);        // ASYNCcnt += 8
  issue_chunk(1, KC);       // ASYNCcnt += 8

  for (int i = 0; i < NKC; ++i) {
    const int b = i & 1;
    // retire chunk i (older group); chunk i+1 may stay in flight (in-order completion)
    if (i < NKC - 1) wait_async_le8(); else wait_async_0();
    __syncthreads();                   // all waves' chunk-i copies visible in LDS

    v16h fah[2], fal[2], fbh[4], fbl[4];
#pragma unroll
    for (int m = 0; m < 2; ++m) {
      const int r = 32 * wm + 16 * m + qn;
      fah[m] = ld_frag(&Ah[b][r][8 * g]);
      fal[m] = ld_frag(&Al[b][r][8 * g]);
    }
#pragma unroll
    for (int j = 0; j < 4; ++j) {
      const int r = 64 * wn + 16 * j + qn;
      fbh[j] = ld_frag(&Bh[b][r][8 * g]);
      fbl[j] = ld_frag(&Bl[b][r][8 * g]);
    }
#pragma unroll
    for (int m = 0; m < 2; ++m) {
#pragma unroll
      for (int j = 0; j < 4; ++j) {
        v8f c = acc[m][j];
        c = wmma16(fal[m], fbh[j], c);
        c = wmma16(fah[m], fbl[j], c);
        c = wmma16(fah[m], fbh[j], c);
        acc[m][j] = c;
      }
    }
    __syncthreads();                   // all waves done reading buffer b
    if (i + 2 < NKC) issue_chunk(b, (i + 2) * KC);
  }

  // ---- per-row max + global atomic max ----
#pragma unroll
  for (int m = 0; m < 2; ++m) {
    v8f mx = acc[m][0];
#pragma unroll
    for (int j = 1; j < 4; ++j)
#pragma unroll
      for (int p = 0; p < 8; ++p) mx[p] = fmaxf(mx[p], acc[m][j][p]);
#pragma unroll
    for (int p = 0; p < 8; ++p) {
      float v = mx[p];
      v = fmaxf(v, __shfl_xor(v, 1, 32));
      v = fmaxf(v, __shfl_xor(v, 2, 32));
      v = fmaxf(v, __shfl_xor(v, 4, 32));
      v = fmaxf(v, __shfl_xor(v, 8, 32));
      if (qn == 0) {
        int row = rowA0 + 32 * wm + 16 * m + 8 * g + p;
        atomicMax(&rowmax[row], encode_f32(v));
      }
    }
  }
}

// ---------------------------------------------------------------------------
// PATH B (small workspace fallback): fused normalize/convert GEMM (round-2 kernel)
// ---------------------------------------------------------------------------
__global__ void norm_kernel(const float* __restrict__ ex,
                            const float* __restrict__ ey,
                            float* __restrict__ invn,
                            unsigned* __restrict__ rowmax,
                            int Nx, int Ny) {
  int b = blockIdx.x;
  const float* src;
  float* dst;
  int row;
  if (b < Nx) { src = ex; row = b;      dst = invn; }
  else        { src = ey; row = b - Nx; dst = invn + Nx; }

  const v4f* p = (const v4f*)(src + (size_t)row * DDIM);
  float ss = 0.0f;
  for (int i = threadIdx.x; i < DDIM / 4; i += blockDim.x) {
    v4f v = p[i];
    ss += v[0]*v[0] + v[1]*v[1] + v[2]*v[2] + v[3]*v[3];
  }
#pragma unroll
  for (int m = 16; m; m >>= 1) ss += __shfl_xor(ss, m, 32);
  __shared__ float red[4];
  int lane = threadIdx.x & 31, wid = threadIdx.x >> 5;
  if (lane == 0) red[wid] = ss;
  __syncthreads();
  if (threadIdx.x == 0) {
    float t = 0.0f;
    for (int i = 0; i < (int)(blockDim.x >> 5); ++i) t += red[i];
    dst[row] = 1.0f / fmaxf(sqrtf(t), 1e-20f);
    if (b < Nx) rowmax[row] = 0u;
  }
}

__global__ __launch_bounds__(256)
void simmax_kernel(const float* __restrict__ ex,
                   const float* __restrict__ ey,
                   const float* __restrict__ invn,
                   unsigned* __restrict__ rowmax,
                   int Nx, int Ny) {
  __shared__ _Float16 Ah[2][BT][LDH];
  __shared__ _Float16 Al[2][BT][LDH];
  __shared__ _Float16 Bh[2][BT][LDH];
  __shared__ _Float16 Bl[2][BT][LDH];
  __shared__ float invA[BT];
  __shared__ float invB[BT];

  const int tid   = threadIdx.x;
  const int rowA0 = blockIdx.y * BT;
  const int rowB0 = blockIdx.x * BT;

  if (tid < BT) invA[tid]      = invn[rowA0 + tid];
  else          invB[tid - BT] = invn[Nx + rowB0 + (tid - BT)];

  const int sr = tid >> 3;
  const int sc = (tid & 7) << 2;

  const int lane = tid & 31;
  const int g    = lane >> 4;
  const int qn   = lane & 15;
  const int wid  = tid >> 5;
  const int wm   = wid & 3;
  const int wn   = wid >> 2;

  v8f acc[2][4] = {};
  v4f ra[4], rb[4];

#pragma unroll
  for (int p = 0; p < 4; ++p) {
    ra[p] = *(const v4f*)(ex + (size_t)(rowA0 + sr + 32 * p) * DDIM + sc);
    rb[p] = *(const v4f*)(ey + (size_t)(rowB0 + sr + 32 * p) * DDIM + sc);
  }
  __syncthreads();

#pragma unroll
  for (int p = 0; p < 4; ++p) {
    const int rr = sr + 32 * p;
    const float ia = invA[rr], ib = invB[rr];
    v4h pah, pal, pbh, pbl;
#pragma unroll
    for (int t = 0; t < 4; ++t) {
      float xa = ra[p][t] * ia;
      _Float16 ha = (_Float16)xa;
      pah[t] = ha; pal[t] = (_Float16)(xa - (float)ha);
      float xb = rb[p][t] * ib;
      _Float16 hb = (_Float16)xb;
      pbh[t] = hb; pbl[t] = (_Float16)(xb - (float)hb);
    }
    *(v4h*)&Ah[0][rr][sc] = pah;  *(v4h*)&Al[0][rr][sc] = pal;
    *(v4h*)&Bh[0][rr][sc] = pbh;  *(v4h*)&Bl[0][rr][sc] = pbl;
  }
#pragma unroll
  for (int p = 0; p < 4; ++p) {
    ra[p] = *(const v4f*)(ex + (size_t)(rowA0 + sr + 32 * p) * DDIM + KC + sc);
    rb[p] = *(const v4f*)(ey + (size_t)(rowB0 + sr + 32 * p) * DDIM + KC + sc);
  }
  __syncthreads();

  for (int i = 0; i < NKC; ++i) {
    const int b = i & 1;
    v16h fah[2], fal[2], fbh[4], fbl[4];
#pragma unroll
    for (int m = 0; m < 2; ++m) {
      const int r = 32 * wm + 16 * m + qn;
      fah[m] = ld_frag(&Ah[b][r][8 * g]);
      fal[m] = ld_frag(&Al[b][r][8 * g]);
    }
#pragma unroll
    for (int j = 0; j < 4; ++j) {
      const int r = 64 * wn + 16 * j + qn;
      fbh[j] = ld_frag(&Bh[b][r][8 * g]);
      fbl[j] = ld_frag(&Bl[b][r][8 * g]);
    }
#pragma unroll
    for (int m = 0; m < 2; ++m) {
#pragma unroll
      for (int j = 0; j < 4; ++j) {
        v8f c = acc[m][j];
        c = wmma16(fal[m], fbh[j], c);
        c = wmma16(fah[m], fbl[j], c);
        c = wmma16(fah[m], fbh[j], c);
        acc[m][j] = c;
      }
    }

    if (i + 1 < NKC) {
      const int nb = b ^ 1;
#pragma unroll
      for (int p = 0; p < 4; ++p) {
        const int rr = sr + 32 * p;
        const float ia = invA[rr], ib = invB[rr];
        v4h pah, pal, pbh, pbl;
#pragma unroll
        for (int t = 0; t < 4; ++t) {
          float xa = ra[p][t] * ia;
          _Float16 ha = (_Float16)xa;
          pah[t] = ha; pal[t] = (_Float16)(xa - (float)ha);
          float xb = rb[p][t] * ib;
          _Float16 hb = (_Float16)xb;
          pbh[t] = hb; pbl[t] = (_Float16)(xb - (float)hb);
        }
        *(v4h*)&Ah[nb][rr][sc] = pah;  *(v4h*)&Al[nb][rr][sc] = pal;
        *(v4h*)&Bh[nb][rr][sc] = pbh;  *(v4h*)&Bl[nb][rr][sc] = pbl;
      }
      if (i + 2 < NKC) {
        const int kc = (i + 2) * KC;
#pragma unroll
        for (int p = 0; p < 4; ++p) {
          ra[p] = *(const v4f*)(ex + (size_t)(rowA0 + sr + 32 * p) * DDIM + kc + sc);
          rb[p] = *(const v4f*)(ey + (size_t)(rowB0 + sr + 32 * p) * DDIM + kc + sc);
        }
      }
    }
    __syncthreads();
  }

#pragma unroll
  for (int m = 0; m < 2; ++m) {
    v8f mx = acc[m][0];
#pragma unroll
    for (int j = 1; j < 4; ++j)
#pragma unroll
      for (int p = 0; p < 8; ++p) mx[p] = fmaxf(mx[p], acc[m][j][p]);
#pragma unroll
    for (int p = 0; p < 8; ++p) {
      float v = mx[p];
      v = fmaxf(v, __shfl_xor(v, 1, 32));
      v = fmaxf(v, __shfl_xor(v, 2, 32));
      v = fmaxf(v, __shfl_xor(v, 4, 32));
      v = fmaxf(v, __shfl_xor(v, 8, 32));
      if (qn == 0) {
        int row = rowA0 + 32 * wm + 16 * m + 8 * g + p;
        atomicMax(&rowmax[row], encode_f32(v));
      }
    }
  }
}

// ---------------------------------------------------------------------------
// Final: lp = log(2)-log(sigma)-0.5*log(2*pi) - (1-cmax)^2/(2*sigma^2); sum.
// ---------------------------------------------------------------------------
__global__ void lp_kernel(const unsigned* __restrict__ rowmax,
                          float* __restrict__ out, int Nx) {
  const float LOG_CONST = 0.97818145169f;
  const float INV_2S2   = 5.5555555556f;
  float s = 0.0f;
  for (int r = threadIdx.x; r < Nx; r += blockDim.x) {
    float cmax = decode_f32(rowmax[r]);
    float x = 1.0f - cmax;
    s += LOG_CONST - x * x * INV_2S2;
  }
#pragma unroll
  for (int m = 16; m; m >>= 1) s += __shfl_xor(s, m, 32);
  __shared__ float red[8];
  int lane = threadIdx.x & 31, wid = threadIdx.x >> 5;
  if (lane == 0) red[wid] = s;
  __syncthreads();
  if (threadIdx.x == 0) {
    float t = 0.0f;
    for (int i = 0; i < 8; ++i) t += red[i];
    out[0] = t;
  }
}

// ---------------------------------------------------------------------------
extern "C" void kernel_launch(void* const* d_in, const int* in_sizes, int n_in,
                              void* d_out, int out_size, void* d_ws, size_t ws_size,
                              hipStream_t stream) {
  (void)n_in; (void)out_size;
  const float* ex = (const float*)d_in[0];
  const float* ey = (const float*)d_in[1];
  const int Nx = in_sizes[0] / DDIM;   // 8192
  const int Ny = in_sizes[1] / DDIM;   // 8192

  const size_t halves = (size_t)DDIM * 2;                 // bytes per row of f16
  const size_t needA  = (size_t)Nx * 4                    // rowmax
                      + ((size_t)Nx + Ny) * halves * 2;   // Xh/Xl + Yh/Yl

  if (ws_size >= needA) {
    // ---- PATH A: precomputed hi/lo f16 + async global->LDS staging ----
    unsigned*  rowmax = (unsigned*)d_ws;
    _Float16*  Xh = (_Float16*)((char*)d_ws + (size_t)Nx * 4);
    _Float16*  Xl = Xh + (size_t)Nx * DDIM;
    _Float16*  Yh = Xl + (size_t)Nx * DDIM;
    _Float16*  Yl = Yh + (size_t)Ny * DDIM;

    prep_kernel<<<Nx + Ny, 128, 0, stream>>>(ex, ey, Xh, Xl, Yh, Yl, rowmax, Nx, Ny);
    dim3 grid(Ny / BT, Nx / BT);
    simmax_async_kernel<<<grid, 256, 0, stream>>>(Xh, Xl, Yh, Yl, rowmax, Nx, Ny);
    lp_kernel<<<1, 256, 0, stream>>>(rowmax, (float*)d_out, Nx);
  } else {
    // ---- PATH B: fused normalize/convert GEMM (small workspace) ----
    float*    invn   = (float*)d_ws;
    unsigned* rowmax = (unsigned*)((char*)d_ws + (size_t)(Nx + Ny) * 4);

    norm_kernel<<<Nx + Ny, 128, 0, stream>>>(ex, ey, invn, rowmax, Nx, Ny);
    dim3 grid(Ny / BT, Nx / BT);
    simmax_kernel<<<grid, 256, 0, stream>>>(ex, ey, invn, rowmax, Nx, Ny);
    lp_kernel<<<1, 256, 0, stream>>>(rowmax, (float*)d_out, Nx);
  }
}
